// DQNSolverCNN6_73031623901429
// MI455X (gfx1250) — matrix operand order repaired and checked by reading, not compiled, not run on hardware
//
#include <hip/hip_runtime.h>
#include <hip/hip_bf16.h>

typedef __attribute__((ext_vector_type(16))) _Float16 v16h;
typedef __attribute__((ext_vector_type(8)))  float    v8f;

#define BATCH   2048
#define DMODEL  128
#define NTOK    81
#define MPAD    96
#define NSTEPS  16
#define ACT_THRESH 0.99f

// f16 weight workspace layout (element offsets)
#define WSOFF_QKV 0        // 384*128 = 49152
#define WSOFF_O   49152    // 128*128 = 16384
#define WSOFF_F1  65536    // 512*128 = 65536
#define WSOFF_F2  131072   // 128*512 = 65536
#define WS_HALVES 196608

struct Params {
  const float* x;
  const float* W_row; const float* b_row;
  const float* W_col; const float* b_col;
  const float* W_box; const float* b_box;
  const float* W_red; const float* b_red;
  const float* bn_g;  const float* bn_b; const float* bn_m; const float* bn_v;
  const float* W_qkv; const float* b_qkv;
  const float* W_o;   const float* b_o;
  const float* ln1_g; const float* ln1_b;
  const float* W_f1;  const float* b_f1;
  const float* W_f2;  const float* b_f2;
  const float* ln2_g; const float* ln2_b;
  const float* W_h;   const float* b_h;
  const float* W_fc;  const float* b_fc;
  const _Float16* Wh;   // f16 weight workspace (may be null in f32 mode)
  float* out;
};

__device__ inline v8f vzero8() {
  v8f z;
#pragma unroll
  for (int i = 0; i < 8; ++i) z[i] = 0.0f;
  return z;
}

__device__ inline v8f wmma_f16(v16h a, v16h b, v8f c) {
  return __builtin_amdgcn_wmma_f32_16x16x32_f16(
      false, a, false, b, (short)0, c, false, false);
}

// A fragment (16x32 f16, row-major stride lda), contiguous K layout:
// lanes 0-15: row=lane, K 0..15; lanes 16-31: row=lane-16, K 16..31
__device__ inline v16h frag_a_lds(const _Float16* base, int lda) {
  int lane = threadIdx.x & 31;
  int row  = lane & 15;
  int koff = (lane >> 4) * 16;
  const _Float16* p = base + row * lda + koff;
  v16h a;
#pragma unroll
  for (int i = 0; i < 8; ++i) a[i] = p[i];
#pragma unroll
  for (int i = 0; i < 8; ++i) a[8 + i] = p[8 + i];
  return a;
}

// B fragment where B[k][n] = W[n][k] (y = x @ W^T).
// lanes 0-15: N=lane, K=0..15 ; lanes 16-31: N=lane-16, K=16..31
template <bool F16W>
__device__ inline v16h frag_b_w(const float* Wf, const _Float16* Wh,
                                int off, int ldw) {
  int lane = threadIdx.x & 31;
  int n  = lane & 15;
  int kb = (lane >> 4) * 16;
  v16h b;
  if constexpr (F16W) {
    const _Float16* p = Wh + off + n * ldw + kb;   // 32 contiguous bytes/lane
#pragma unroll
    for (int i = 0; i < 16; ++i) b[i] = p[i];
  } else {
    const float* p = Wf + off + n * ldw + kb;
#pragma unroll
    for (int i = 0; i < 16; ++i) b[i] = (_Float16)p[i];
  }
  return b;
}

// B fragment where B[k][n] = M[n][k], M f16 row-major stride ldm (scores = Q K^T)
__device__ inline v16h frag_b_T_h(const _Float16* M, int ldm) {
  int lane = threadIdx.x & 31;
  int n  = lane & 15;
  int kb = (lane >> 4) * 16;
  const _Float16* p = M + n * ldm + kb;
  v16h b;
#pragma unroll
  for (int i = 0; i < 16; ++i) b[i] = p[i];
  return b;
}

// B fragment where B[k][n] = M[k][n] (attn @ V), M f16 row-major stride ldm
__device__ inline v16h frag_b_h(const _Float16* M, int ldm) {
  int lane = threadIdx.x & 31;
  int n  = lane & 15;
  int kb = (lane >> 4) * 16;
  const _Float16* p = M + kb * ldm + n;
  v16h b;
#pragma unroll
  for (int i = 0; i < 16; ++i) b[i] = p[i * ldm];
  return b;
}

__device__ inline void layernorm_rows(const float* src, _Float16* dst,
                                      const float* g, const float* bb, int tid) {
  if (tid < MPAD) {
    int m = tid;
    float mean = 0.f;
    for (int d = 0; d < DMODEL; ++d) mean += src[m * DMODEL + d];
    mean *= (1.0f / DMODEL);
    float var = 0.f;
    for (int d = 0; d < DMODEL; ++d) {
      float t = src[m * DMODEL + d] - mean;
      var += t * t;
    }
    var *= (1.0f / DMODEL);
    float inv = rsqrtf(var + 1e-5f);
    if (m < NTOK) {
      for (int d = 0; d < DMODEL; ++d)
        dst[m * DMODEL + d] =
            (_Float16)((src[m * DMODEL + d] - mean) * inv * g[d] + bb[d]);
    } else {
      for (int d = 0; d < DMODEL; ++d) dst[m * DMODEL + d] = (_Float16)0.f;
    }
  }
}

__global__ void convert_weights_f16(const float* Wqkv, const float* Wo,
                                    const float* Wf1, const float* Wf2,
                                    _Float16* ws) {
  int i = blockIdx.x * 256 + threadIdx.x;          // grid covers 65536
  if (i < 49152) ws[WSOFF_QKV + i] = (_Float16)Wqkv[i];
  if (i < 16384) ws[WSOFF_O   + i] = (_Float16)Wo[i];
  if (i < 65536) {
    ws[WSOFF_F1 + i] = (_Float16)Wf1[i];
    ws[WSOFF_F2 + i] = (_Float16)Wf2[i];
  }
}

template <bool F16W>
__global__ __launch_bounds__(256, 1)
void sudoku_hrm_fused(Params p) {
  // ---- static LDS layout (211 KB; fits 320 KB/WGP) ----
  __shared__ __align__(16) unsigned char smem[216128];
  _Float16* sT   = (_Float16*)(smem + 0);        // 96x128 f16 token state
  _Float16* sQ   = (_Float16*)(smem + 24576);    // 96x128 f16
  _Float16* sK   = (_Float16*)(smem + 49152);    // 96x128 f16
  _Float16* sV   = (_Float16*)(smem + 73728);    // 96x128 f16
  _Float16* sO   = (_Float16*)(smem + 98304);    // 96x128 f16 attention out
  _Float16* sP   = (_Float16*)(smem + 122880);   // 96x96 probs
  float*    sR   = (float*)(smem + 141312);      // 96x128 f32 resid / 96x96 scores
  _Float16* sFF  = (_Float16*)(smem + 190464);   // 96x128 f16 FFN chunk
  float*    sRed = (float*)(smem + 215040);      // 256 f32 reduction
  float*    sScal= (float*)(smem + 216064);      // halt, ponder, sp
  // init-phase overlays inside sR region (disjoint lifetimes)
  float* sX  = (float*)(smem + 141312);          // 10*81 floats
  float* sFr = sX + 832;                         // 48*9
  float* sFc = sFr + 432;                        // 48*9
  float* sFb = sFc + 432;                        // 48*9

  const int tid  = threadIdx.x;
  const int wave = tid >> 5;
  const int lane = tid & 31;
  const int col  = lane & 15;
  const int rb   = (lane >> 4) * 8;
  const int b    = blockIdx.x;

  // ================= Phase 0: load x tile =================
  for (int i = tid; i < 810; i += 256) sX[i] = p.x[(size_t)b * 810 + i];
  if (tid == 0) { sScal[0] = 0.f; sScal[1] = 0.f; sScal[2] = 0.f; }
  __syncthreads();

  // ================= Phase 1: constraint convs =================
  for (int idx = tid; idx < 1296; idx += 256) {
    int which = idx / 432;
    int rest  = idx - which * 432;
    int f = rest / 9, pos = rest % 9;
    if (which == 0) {            // row conv (1x9)
      float acc = p.b_row[f];
      for (int ci = 0; ci < 10; ++ci)
        for (int j = 0; j < 9; ++j)
          acc += sX[ci * 81 + pos * 9 + j] * p.W_row[f * 90 + ci * 9 + j];
      sFr[f * 9 + pos] = acc;
    } else if (which == 1) {     // col conv (9x1)
      float acc = p.b_col[f];
      for (int ci = 0; ci < 10; ++ci)
        for (int i = 0; i < 9; ++i)
          acc += sX[ci * 81 + i * 9 + pos] * p.W_col[f * 90 + ci * 9 + i];
      sFc[f * 9 + pos] = acc;
    } else {                     // box conv (3x3 stride 3)
      int br = pos / 3, bc = pos % 3;
      float acc = p.b_box[f];
      for (int ci = 0; ci < 10; ++ci)
        for (int di = 0; di < 3; ++di)
          for (int dj = 0; dj < 3; ++dj)
            acc += sX[ci * 81 + (br * 3 + di) * 9 + (bc * 3 + dj)] *
                   p.W_box[f * 90 + ci * 9 + di * 3 + dj];
      sFb[f * 9 + pos] = acc;
    }
  }
  __syncthreads();

  // ================= Phase 2: reduce 144->128 + BN + ReLU -> state ===========
  for (int idx = tid; idx < MPAD * DMODEL; idx += 256) {
    int m = idx >> 7, d = idx & 127;
    float val = 0.f;
    if (m < NTOK) {
      int r = m / 9, c = m % 9, bx = (r / 3) * 3 + (c / 3);
      const float* wr = p.W_red + d * 144;
      float acc = p.b_red[d];
      for (int k = 0; k < 48; ++k) acc += wr[k]      * sFr[k * 9 + r];
      for (int k = 0; k < 48; ++k) acc += wr[48 + k] * sFc[k * 9 + c];
      for (int k = 0; k < 48; ++k) acc += wr[96 + k] * sFb[k * 9 + bx];
      acc = (acc - p.bn_m[d]) * rsqrtf(p.bn_v[d] + 1e-5f) * p.bn_g[d] + p.bn_b[d];
      val = fmaxf(acc, 0.f);
    }
    sT[m * DMODEL + d] = (_Float16)val;
  }

  // q-head accumulators (ssum folded through the linear head)
  float qv0 = 0.f, qv1 = 0.f, qv2 = 0.f;

  // ================= ACT loop =================
  for (int step = 0; step < NSTEPS; ++step) {
    __syncthreads();
    float haltCur = sScal[0];
    if (haltCur >= ACT_THRESH) continue;   // uniform per block

    // -------- full QKV projection: 24 N-tiles x 6 M-tiles, K=128 --------
    // wave owns 3 N-tiles; B fragments hoisted and reused over 6 M-tiles
    for (int c = 0; c < 3; ++c) {
      int nt = wave * 3 + c;               // 0..23 over 384 output cols
      int wrow = nt * 16;
      v16h bw[4];
#pragma unroll
      for (int kt = 0; kt < 4; ++kt)
        bw[kt] = frag_b_w<F16W>(p.W_qkv, p.Wh + WSOFF_QKV,
                                wrow * DMODEL + kt * 32, DMODEL);
      _Float16* dst = (nt < 8) ? sQ : (nt < 16) ? sK : sV;
      int dcol = (nt & 7) * 16;
      float bias = p.b_qkv[wrow + col];
      for (int mt = 0; mt < 6; ++mt) {
        v8f acc = vzero8();
#pragma unroll
        for (int kt = 0; kt < 4; ++kt) {
          v16h a = frag_a_lds(sT + mt * 16 * DMODEL + kt * 32, DMODEL);
          acc = wmma_f16(a, bw[kt], acc);
        }
#pragma unroll
        for (int i = 0; i < 8; ++i)
          dst[(mt * 16 + rb + i) * DMODEL + dcol + col] = (_Float16)(acc[i] + bias);
      }
    }
    __syncthreads();

    // -------- attention per head --------
    for (int h = 0; h < 4; ++h) {
      // scores = Q_h @ K_h^T / sqrt(32): 6x6 tiles, K=32
      for (int t = wave; t < 36; t += 8) {
        int mt = t / 6, nt = t % 6;
        v16h a  = frag_a_lds(sQ + mt * 16 * DMODEL + h * 32, DMODEL);
        v16h bf = frag_b_T_h(sK + nt * 16 * DMODEL + h * 32, DMODEL);
        v8f acc = vzero8();
        acc = wmma_f16(a, bf, acc);
#pragma unroll
        for (int i = 0; i < 8; ++i)
          sR[(mt * 16 + rb + i) * 96 + nt * 16 + col] =
              acc[i] * 0.17677669529663687f;
      }
      __syncthreads();

      // softmax over 81 valid keys; zero padded columns
      if (tid < MPAD) {
        int m = tid;
        float mx = -1e30f;
        for (int j = 0; j < NTOK; ++j) mx = fmaxf(mx, sR[m * 96 + j]);
        float s = 0.f;
        for (int j = 0; j < NTOK; ++j) {
          float e = __expf(sR[m * 96 + j] - mx);
          sR[m * 96 + j] = e;
          s += e;
        }
        float inv = 1.f / s;
        for (int j = 0; j < NTOK; ++j) sP[m * 96 + j] = (_Float16)(sR[m * 96 + j] * inv);
        for (int j = NTOK; j < 96; ++j) sP[m * 96 + j] = (_Float16)0.f;
      }
      __syncthreads();

      // out_h = P @ V_h : 6x2 tiles, K=96
      for (int t = wave; t < 12; t += 8) {
        int mt = t / 2, nt = t % 2;
        v8f acc = vzero8();
#pragma unroll
        for (int kt = 0; kt < 3; ++kt) {
          v16h a  = frag_a_lds(sP + mt * 16 * 96 + kt * 32, 96);
          v16h bf = frag_b_h(sV + kt * 32 * DMODEL + h * 32 + nt * 16, DMODEL);
          acc = wmma_f16(a, bf, acc);
        }
#pragma unroll
        for (int i = 0; i < 8; ++i)
          sO[(mt * 16 + rb + i) * DMODEL + h * 32 + nt * 16 + col] = (_Float16)acc[i];
      }
      __syncthreads();
    }

    // -------- out projection + residual: wave owns N-tile `wave` --------
    {
      v16h bo[4];
#pragma unroll
      for (int kt = 0; kt < 4; ++kt)
        bo[kt] = frag_b_w<F16W>(p.W_o, p.Wh + WSOFF_O,
                                (wave * 16) * DMODEL + kt * 32, DMODEL);
      int n = wave * 16 + col;
      float bias = p.b_o[n];
      for (int mt = 0; mt < 6; ++mt) {
        v8f acc = vzero8();
#pragma unroll
        for (int kt = 0; kt < 4; ++kt) {
          v16h a = frag_a_lds(sO + mt * 16 * DMODEL + kt * 32, DMODEL);
          acc = wmma_f16(a, bo[kt], acc);
        }
#pragma unroll
        for (int i = 0; i < 8; ++i) {
          int row = mt * 16 + rb + i;
          sR[row * DMODEL + n] = acc[i] + bias + (float)sT[row * DMODEL + n];
        }
      }
    }
    __syncthreads();
    layernorm_rows(sR, sT, p.ln1_g, p.ln1_b, tid);
    __syncthreads();

    // -------- FFN: 128 -> 512 -> 128, streamed in 4 K-chunks --------
    v8f facc[6];
#pragma unroll
    for (int j = 0; j < 6; ++j) facc[j] = vzero8();

    for (int cc = 0; cc < 4; ++cc) {
      // ff1 chunk: relu(t2 @ W_f1[cc]^T + b); wave owns N-tile `wave`
      {
        v16h b1[4];
#pragma unroll
        for (int kt = 0; kt < 4; ++kt)
          b1[kt] = frag_b_w<F16W>(p.W_f1, p.Wh + WSOFF_F1,
                                  (cc * DMODEL + wave * 16) * DMODEL + kt * 32,
                                  DMODEL);
        int n = wave * 16 + col;
        float bias = p.b_f1[cc * DMODEL + n];
        for (int mt = 0; mt < 6; ++mt) {
          v8f acc = vzero8();
#pragma unroll
          for (int kt = 0; kt < 4; ++kt) {
            v16h a = frag_a_lds(sT + mt * 16 * DMODEL + kt * 32, DMODEL);
            acc = wmma_f16(a, b1[kt], acc);
          }
#pragma unroll
          for (int i = 0; i < 8; ++i)
            sFF[(mt * 16 + rb + i) * DMODEL + n] =
                (_Float16)fmaxf(acc[i] + bias, 0.f);
        }
      }
      __syncthreads();

      // ff2 partial accumulate: wave owns N-tile `wave`, B hoisted per K-tile
#pragma unroll
      for (int kt = 0; kt < 4; ++kt) {
        v16h b2 = frag_b_w<F16W>(p.W_f2, p.Wh + WSOFF_F2,
                                 (wave * 16) * 512 + cc * DMODEL + kt * 32, 512);
#pragma unroll
        for (int j = 0; j < 6; ++j) {
          v16h a = frag_a_lds(sFF + j * 16 * DMODEL + kt * 32, DMODEL);
          facc[j] = wmma_f16(a, b2, facc[j]);
        }
      }
      __syncthreads();
    }

    // residual + store ff2 result
    {
      int n = wave * 16 + col;
      float bias = p.b_f2[n];
#pragma unroll
      for (int j = 0; j < 6; ++j)
#pragma unroll
        for (int i = 0; i < 8; ++i) {
          int row = j * 16 + rb + i;
          sR[row * DMODEL + n] = facc[j][i] + bias + (float)sT[row * DMODEL + n];
        }
    }
    __syncthreads();
    layernorm_rows(sR, sT, p.ln2_g, p.ln2_b, tid);
    __syncthreads();

    // -------- ACT halting --------
    float part = 0.f;
    for (int idx = tid; idx < NTOK * DMODEL; idx += 256) {
      int m = idx >> 7, d = idx & 127;
      part += (float)sT[m * DMODEL + d] * p.W_h[d];
    }
    sRed[tid] = part;
    __syncthreads();
    for (int off = 128; off > 0; off >>= 1) {
      if (tid < off) sRed[tid] += sRed[tid + off];
      __syncthreads();
    }
    if (tid == 0) {
      float gf = sRed[0] * (1.0f / 81.0f);
      float hp = 1.f / (1.f + __expf(-(gf + p.b_h[0])));
      float halt = sScal[0];
      float sp = fminf(hp, 1.f - halt);
      float nh = halt + sp;
      sScal[0] = nh;
      sScal[1] += (nh < 1.f) ? 1.f : 0.f;
      sScal[2] = sp;
    }
    __syncthreads();

    // -------- accumulate q-head: qv += sp * (ns @ W_fc^T) --------
    {
      float sp = sScal[2];
#pragma unroll
      for (int r = 0; r < 3; ++r) {
        int t = tid + r * 256;
        if (t < 729) {
          int m = t / 9, jj = t % 9;
          float acc = 0.f;
          for (int d = 0; d < DMODEL; ++d)
            acc += (float)sT[m * DMODEL + d] * p.W_fc[jj * DMODEL + d];
          float v = sp * acc;
          if (r == 0) qv0 += v; else if (r == 1) qv1 += v; else qv2 += v;
        }
      }
    }
  }

  // ================= finalize: remainder term + outputs =================
  __syncthreads();
  float rem = 1.f - sScal[0];
#pragma unroll
  for (int r = 0; r < 3; ++r) {
    int t = tid + r * 256;
    if (t < 729) {
      int m = t / 9, jj = t % 9;
      float acc = 0.f;
      for (int d = 0; d < DMODEL; ++d)
        acc += (float)sT[m * DMODEL + d] * p.W_fc[jj * DMODEL + d];
      float qbase = (r == 0) ? qv0 : (r == 1) ? qv1 : qv2;
      p.out[(size_t)b * 729 + t] = qbase + rem * acc + p.b_fc[jj];
    }
  }
  if (tid == 0) p.out[(size_t)BATCH * 729 + b] = sScal[1] + rem;
}

extern "C" void kernel_launch(void* const* d_in, const int* in_sizes, int n_in,
                              void* d_out, int out_size, void* d_ws, size_t ws_size,
                              hipStream_t stream) {
  (void)in_sizes; (void)n_in; (void)out_size;
  Params p;
  p.x     = (const float*)d_in[0];
  p.W_row = (const float*)d_in[1];  p.b_row = (const float*)d_in[2];
  p.W_col = (const float*)d_in[3];  p.b_col = (const float*)d_in[4];
  p.W_box = (const float*)d_in[5];  p.b_box = (const float*)d_in[6];
  p.W_red = (const float*)d_in[7];  p.b_red = (const float*)d_in[8];
  p.bn_g  = (const float*)d_in[9];  p.bn_b  = (const float*)d_in[10];
  p.bn_m  = (const float*)d_in[11]; p.bn_v  = (const float*)d_in[12];
  p.W_qkv = (const float*)d_in[13]; p.b_qkv = (const float*)d_in[14];
  p.W_o   = (const float*)d_in[15]; p.b_o   = (const float*)d_in[16];
  p.ln1_g = (const float*)d_in[17]; p.ln1_b = (const float*)d_in[18];
  p.W_f1  = (const float*)d_in[19]; p.b_f1  = (const float*)d_in[20];
  p.W_f2  = (const float*)d_in[21]; p.b_f2  = (const float*)d_in[22];
  p.ln2_g = (const float*)d_in[23]; p.ln2_b = (const float*)d_in[24];
  p.W_h   = (const float*)d_in[25]; p.b_h   = (const float*)d_in[26];
  p.W_fc  = (const float*)d_in[27]; p.b_fc  = (const float*)d_in[28];
  p.out   = (float*)d_out;

  bool f16w = ws_size >= (size_t)WS_HALVES * sizeof(_Float16);
  if (f16w) {
    _Float16* wsh = (_Float16*)d_ws;
    p.Wh = wsh;
    convert_weights_f16<<<256, 256, 0, stream>>>(p.W_qkv, p.W_o, p.W_f1,
                                                 p.W_f2, wsh);
    sudoku_hrm_fused<true><<<BATCH, 256, 0, stream>>>(p);
  } else {
    p.Wh = nullptr;
    sudoku_hrm_fused<false><<<BATCH, 256, 0, stream>>>(p);
  }
}
